// FraudDetectionHGNN_55516747268374
// MI455X (gfx1250) — matrix operand order repaired and checked by reading, not compiled, not run on hardware
//
#include <hip/hip_runtime.h>
#include <hip/hip_bf16.h>

// ---------------------------------------------------------------------------
// CDNA5 WMMA types (wave32, v_wmma_f32_16x16x32_bf16)
// ---------------------------------------------------------------------------
typedef __attribute__((ext_vector_type(16))) __bf16 v16bf;
typedef __attribute__((ext_vector_type(8)))  float  v8f;

#define NTOT_C 220000

// ---------------------------------------------------------------------------
// Fragment layouts (CDNA5 ISA 7.12.2), bf16 16x16x32:
//  A (16x32 MxK): lane l: row=l&15; j in [0,8):  K=(l>>4)*8+j
//                                   j in [8,16): K=16+(l>>4)*8+(j-8)
//    -> each half-fragment (8 elems, 16B) is 8 K-contiguous bf16 of one row.
//  B (32x16 KxN): lane l: col=l&15; K as for A: j<8: (l>>4)*8+j; j>=8: 16+(l>>4)*8+(j-8)
//  C/D (16x16 f32): elem r: m = r + 8*(lane>>4); n = lane&15
// LDS slot arrays: [tile][lane][16 elems] -> 32B contiguous per lane fragment.
// ---------------------------------------------------------------------------
__device__ __forceinline__ int b_slot(int nn, int k5) {
    return (nn + (((k5 >> 3) & 1) << 4)) * 16 + ((k5 >> 4) << 3) + (k5 & 7);
}

// Async global->LDS 16-byte copy (GLOBAL_LOAD_ASYNC_TO_LDS_B128, ASYNCcnt).
__device__ __forceinline__ void async_cp16(const void* lds_dst, const void* gsrc) {
    unsigned lo = (unsigned)(unsigned long long)lds_dst;  // low 32b = LDS offset
    asm volatile("global_load_async_to_lds_b128 %0, %1, off"
                 :: "v"(lo), "v"(gsrc) : "memory");
}
__device__ __forceinline__ void async_wait0() {
#if __has_builtin(__builtin_amdgcn_s_wait_asynccnt)
    __builtin_amdgcn_s_wait_asynccnt(0);
#else
    asm volatile("s_wait_asynccnt 0x0" ::: "memory");
#endif
}

// ---------------------------------------------------------------------------
// GEMM: out[M x 128] = epi(A[M x K](bf16) @ W[K x 128](f32) + bias)
// Block = 128x128 tile, 256 threads = 8 waves (wave w -> rows [w*16,+16)).
// A staged via async-to-LDS (fragment shuffle folded into per-lane addresses);
// W staged fp32->bf16 into fragment layout by VALU.
// Main loop: per K-step, 8 INDEPENDENT accumulator chains (one per N-tile)
// so the XDL pipe always has non-dependent WMMAs to issue.
// EPI 0: o = acc + bias
// EPI 1: o = leaky_relu(gate*(acc+bias) + (1-gate)*Hprev), gate = sigmoid(*skip)
// ---------------------------------------------------------------------------
template<int EPI, int KT>
__global__ __launch_bounds__(256) void gemm_n128(
    const __bf16* __restrict__ A, int lda,      // bf16 activations, padded rows
    const float* __restrict__ W,                // K x 128 row-major fp32
    const float* __restrict__ bias,             // 128
    float*  __restrict__ C,                     // M x 128 (nullable)
    __bf16* __restrict__ Cb,                    // M x 128 (nullable)
    const float* __restrict__ skipGate,
    const float* __restrict__ Hprev,            // M x 128 (EPI==1)
    int M, int K)
{
    __shared__ __attribute__((aligned(32))) __bf16 ldsA[8 * KT * 512];
    __shared__ __attribute__((aligned(32))) __bf16 ldsB[8 * KT * 512];

    const int tid = threadIdx.x;
    const int M0  = blockIdx.x * 128;

    // ---- A: async copy, fragment shuffle in addresses (16B per lane-chunk) ----
    for (int c = tid; c < 8 * KT * 64; c += 256) {
        int tile = c >> 6;          // mt*KT + kt
        int l    = (c >> 1) & 31;   // fragment lane
        int half = c & 1;           // j 0..7 vs 8..15
        int mt = tile / KT, kt = tile - mt * KT;
        int row = (mt << 4) + (l & 15);
        int kk  = (kt << 5) + (half << 4) + ((l >> 4) << 3);
        const __bf16* g = A + (size_t)(M0 + row) * lda + kk;
        async_cp16(&ldsA[tile * 512 + l * 16 + half * 8], g);
    }
    // ---- W: fp32 load, convert, fragment-layout LDS store ----
    for (int idx = tid; idx < KT * 32 * 128; idx += 256) {
        int k = idx >> 7, n = idx & 127;
        float v = (k < K) ? W[(size_t)k * 128 + n] : 0.0f;
        int nt = n >> 4, nn = n & 15;
        int kt = k >> 5, k5 = k & 31;
        ldsB[((nt * KT) + kt) * 512 + b_slot(nn, k5)] = (__bf16)v;
    }
    async_wait0();
    __syncthreads();

    const int wave = tid >> 5, lane = tid & 31;
    const int mt = wave;

    v16bf afrag[KT];
#pragma unroll
    for (int kt = 0; kt < KT; ++kt)
        afrag[kt] = *(const v16bf*)&ldsA[((mt * KT) + kt) * 512 + lane * 16];

    // 8 independent accumulator chains (one per 16-wide N tile)
    v8f acc[8];
#pragma unroll
    for (int nt = 0; nt < 8; ++nt) acc[nt] = v8f{};

#pragma unroll
    for (int kt = 0; kt < KT; ++kt) {
#pragma unroll
        for (int nt = 0; nt < 8; ++nt) {
            v16bf bfrag = *(const v16bf*)&ldsB[((nt * KT) + kt) * 512 + lane * 16];
            acc[nt] = __builtin_amdgcn_wmma_f32_16x16x32_bf16(
                false, afrag[kt], false, bfrag, (short)0, acc[nt], false, false);
        }
    }

    // ---- epilogue ----
    const int n_in = lane & 15;
    const int m_hi = (lane >> 4) << 3;
    const int rbase = M0 + (mt << 4) + m_hi;

    float gate = 1.0f, ogate = 0.0f;
    if (EPI == 1) {
        float s = skipGate[0];
        gate  = 1.0f / (1.0f + __expf(-s));
        ogate = 1.0f - gate;
    }

    if (M0 + 128 <= M) {                  // uniform fast path: no guards
#pragma unroll
        for (int nt = 0; nt < 8; ++nt) {
            const int col = (nt << 4) + n_in;
            const float b = bias ? bias[col] : 0.0f;
#pragma unroll
            for (int r = 0; r < 8; ++r) {
                float o = acc[nt][r] + b;
                if (EPI == 1) {
                    float hp = Hprev[(size_t)(rbase + r) * 128 + col];
                    o = gate * o + ogate * hp;
                    o = o > 0.0f ? o : 0.01f * o;
                }
                size_t off = (size_t)(rbase + r) * 128 + col;
                if (C)  C[off]  = o;
                if (Cb) Cb[off] = (__bf16)o;
            }
        }
    } else {                              // ragged last block only
#pragma unroll
        for (int nt = 0; nt < 8; ++nt) {
            const int col = (nt << 4) + n_in;
            const float b = bias ? bias[col] : 0.0f;
#pragma unroll
            for (int r = 0; r < 8; ++r) {
                if (rbase + r < M) {
                    float o = acc[nt][r] + b;
                    if (EPI == 1) {
                        float hp = Hprev[(size_t)(rbase + r) * 128 + col];
                        o = gate * o + ogate * hp;
                        o = o > 0.0f ? o : 0.01f * o;
                    }
                    size_t off = (size_t)(rbase + r) * 128 + col;
                    if (C)  C[off]  = o;
                    if (Cb) Cb[off] = (__bf16)o;
                }
            }
        }
    }
}

// ---------------------------------------------------------------------------
// Per-head relation transform: out[n, h*32+e] = sum_d A[n, h*32+d] * Wrel[h,d,e]
// grid.y = head. 128x32 tile, K = 32 (single bf16 WMMA step). A is bf16.
// ---------------------------------------------------------------------------
__global__ __launch_bounds__(256) void relgemm(
    const __bf16* __restrict__ A,    // Nsrc x 128 (bf16, padded rows)
    const float* __restrict__ Wrel,  // 4 x 32 x 32
    float* __restrict__ Cout,        // Nsrc x 128
    int M)
{
    const int h = blockIdx.y;
    __shared__ __attribute__((aligned(32))) __bf16 ldsA[8 * 512];
    __shared__ __attribute__((aligned(32))) __bf16 ldsB[2 * 512];
    const float* Wh = Wrel + h * 32 * 32;
    const int tid = threadIdx.x;
    const int M0  = blockIdx.x * 128;

    for (int c = tid; c < 8 * 64; c += 256) {
        int tile = c >> 6, l = (c >> 1) & 31, half = c & 1;
        int row = (tile << 4) + (l & 15);
        int kk  = (half << 4) + ((l >> 4) << 3);
        const __bf16* g = A + (size_t)(M0 + row) * 128 + h * 32 + kk;
        async_cp16(&ldsA[tile * 512 + l * 16 + half * 8], g);
    }
    for (int idx = tid; idx < 32 * 32; idx += 256) {
        int k5 = idx >> 5, n = idx & 31;
        int nt = n >> 4, nn = n & 15;
        ldsB[nt * 512 + b_slot(nn, k5)] = (__bf16)Wh[k5 * 32 + n];
    }
    async_wait0();
    __syncthreads();

    const int wave = tid >> 5, lane = tid & 31;
    v16bf af = *(const v16bf*)&ldsA[wave * 512 + lane * 16];
    const int n_in = lane & 15;
    const int rbase = M0 + (wave << 4) + ((lane >> 4) << 3);

    // two independent accumulator chains
    v8f acc[2];
#pragma unroll
    for (int nt = 0; nt < 2; ++nt) {
        v16bf bf = *(const v16bf*)&ldsB[nt * 512 + lane * 16];
        acc[nt] = __builtin_amdgcn_wmma_f32_16x16x32_bf16(
            false, af, false, bf, (short)0, v8f{}, false, false);
    }
    if (M0 + 128 <= M) {
#pragma unroll
        for (int nt = 0; nt < 2; ++nt) {
            const int col = h * 32 + (nt << 4) + n_in;
#pragma unroll
            for (int r = 0; r < 8; ++r)
                Cout[(size_t)(rbase + r) * 128 + col] = acc[nt][r];
        }
    } else {
#pragma unroll
        for (int nt = 0; nt < 2; ++nt) {
            const int col = h * 32 + (nt << 4) + n_in;
#pragma unroll
            for (int r = 0; r < 8; ++r)
                if (rbase + r < M)
                    Cout[(size_t)(rbase + r) * 128 + col] = acc[nt][r];
        }
    }
}

// ---------------------------------------------------------------------------
// fp32 x -> bf16, pad rows to Mpad and cols to 32
// ---------------------------------------------------------------------------
__global__ void cvt_pad(const float* __restrict__ x, __bf16* __restrict__ xb,
                        int M, int Kin, int Mpad)
{
    int i = blockIdx.x * blockDim.x + threadIdx.x;
    if (i >= Mpad * 32) return;
    int row = i >> 5, c = i & 31;
    float v = (row < M && c < Kin) ? x[(size_t)row * Kin + c] : 0.0f;
    xb[i] = (__bf16)v;
}

// ---------------------------------------------------------------------------
// Edge phase: score + segment max, exp + segment sum, weighted scatter
// ---------------------------------------------------------------------------
__device__ __forceinline__ int enc_f(float x) {
    int i = __float_as_int(x);
    return i >= 0 ? i : (i ^ 0x7fffffff);
}
__device__ __forceinline__ float dec_f(int i) {
    return __int_as_float(i >= 0 ? i : (i ^ 0x7fffffff));
}

__global__ void attn_score(const int* __restrict__ ei, int nE, int dstGOff,
                           const float* __restrict__ q,
                           const float* __restrict__ keb,
                           const float* __restrict__ prel,
                           float* __restrict__ ae, int* __restrict__ mx)
{
    int i = blockIdx.x * blockDim.x + threadIdx.x;
    if (i >= nE * 4) return;
    int e = i >> 2, h = i & 3;
    int s = ei[e], d = ei[nE + e];
    const float* qp = q   + ((size_t)(dstGOff + d) * 128 + h * 32);
    const float* kp = keb + ((size_t)s * 128 + h * 32);
    float acc = 0.0f;
#pragma unroll
    for (int j = 0; j < 32; j += 4) {
        float4 qa = *(const float4*)(qp + j);
        float4 ka = *(const float4*)(kp + j);
        acc += qa.x * ka.x + qa.y * ka.y + qa.z * ka.z + qa.w * ka.w;
    }
    float a = acc * prel[h] * 0.17677669529663687f; // 1/sqrt(32)
    ae[i] = a;
    atomicMax(&mx[(size_t)(dstGOff + d) * 4 + h], enc_f(a));
}

__global__ void attn_ex(const int* __restrict__ ei, int nE, int dstGOff,
                        const float* __restrict__ ae, const int* __restrict__ mx,
                        float* __restrict__ wex, float* __restrict__ den)
{
    int i = blockIdx.x * blockDim.x + threadIdx.x;
    if (i >= nE * 4) return;
    int e = i >> 2, h = i & 3;
    int d = ei[nE + e];
    float m  = dec_f(mx[(size_t)(dstGOff + d) * 4 + h]);
    float ex = __expf(ae[i] - m);
    wex[i] = ex;
    atomicAdd(&den[(size_t)(dstGOff + d) * 4 + h], ex);
}

__global__ void attn_scatter(const int* __restrict__ ei, int nE, int dstGOff,
                             const float* __restrict__ wex,
                             const float* __restrict__ veb,
                             float* __restrict__ num)
{
    long i = (long)blockIdx.x * blockDim.x + threadIdx.x;
    if (i >= (long)nE * 128) return;
    int e = (int)(i >> 7), j = (int)(i & 127), h = j >> 5;
    int s = ei[e], d = ei[nE + e];
    atomicAdd(&num[(size_t)(dstGOff + d) * 128 + j],
              wex[e * 4 + h] * veb[(size_t)s * 128 + j]);
}

// g(bf16) = gelu(num / (den + 1e-16))
__global__ void gelu_div(const float* __restrict__ num,
                         const float* __restrict__ den,
                         __bf16* __restrict__ g, long n)
{
    long i = (long)blockIdx.x * blockDim.x + threadIdx.x;
    if (i >= n) return;
    int j = (int)(i & 127), h = j >> 5;
    float d = den[(i >> 7) * 4 + h] + 1e-16f;
    float x = num[i] / d;
    g[i] = (__bf16)(0.5f * x * (1.0f + erff(x * 0.7071067811865476f)));
}

__global__ void fill_f(float* __restrict__ p, float v, long n)
{
    long i = (long)blockIdx.x * blockDim.x + threadIdx.x;
    if (i < n) p[i] = v;
}
__global__ void fill_i(int* __restrict__ p, int v, long n)
{
    long i = (long)blockIdx.x * blockDim.x + threadIdx.x;
    if (i < n) p[i] = v;
}

__global__ void out_head(const float* __restrict__ h,
                         const float* __restrict__ Wo,  // 128 x 2
                         const float* __restrict__ bo,  // 2
                         float* __restrict__ out, int M)
{
    int r = blockIdx.x * blockDim.x + threadIdx.x;
    if (r >= M) return;
    float a0 = bo[0], a1 = bo[1];
    const float* hp = h + (size_t)r * 128;
#pragma unroll 4
    for (int j = 0; j < 128; ++j) {
        float x = hp[j];
        a0 += x * Wo[2 * j];
        a1 += x * Wo[2 * j + 1];
    }
    out[2 * r]     = a0;
    out[2 * r + 1] = a1;
}

// ---------------------------------------------------------------------------
// Host orchestration
// ---------------------------------------------------------------------------
extern "C" void kernel_launch(void* const* d_in, const int* in_sizes, int n_in,
                              void* d_out, int out_size, void* d_ws, size_t ws_size,
                              hipStream_t stream)
{
    (void)n_in; (void)out_size; (void)ws_size;

    // Node types in jax-sorted param order: 0=customer, 1=order, 2=product
    const int NN[3]   = {80000, 100000, 40000};
    const int GOFF[3] = {100000, 0, 180000};   // reference OFFS (order first)
    const int INCH[3] = {16, 32, 16};
    const int NTOT = NTOT_C;

    // Edge types sorted by name: cpo, ocp, orc, prc -> input indices 3,5,6,4
    const int EIDX[4]   = {3, 5, 6, 4};
    const int ET_SRC[4] = {0, 1, 1, 2};
    const int ET_DST[4] = {1, 2, 0, 1};
    const int* ei[4];
    int nE[4];
    for (int t = 0; t < 4; ++t) {
        ei[t] = (const int*)d_in[EIDX[t]];
        nE[t] = in_sizes[EIDX[t]] / 2;
    }

    auto F = [&](int i) { return (const float*)d_in[i]; };
    // Param leaves (jax tree_flatten, sorted keys), base = 7:
    //  conv: a(6) a_rel(4) k(6) m_rel(4) p_rel(4) q(6) skip(3) v(6) = 39
    const int P = 7, CONV1 = P, CONV2 = P + 39, EMB = P + 78, OUTP = P + 84;
    const float* X[3] = {F(1), F(0), F(2)};  // sorted: customer, order, product

    // ---- workspace bump allocator (floats, 256B-aligned chunks) ----
    float* WSB = (float*)d_ws;
    size_t cur = 0;
    auto alloc = [&](size_t nf) {
        nf = (nf + 63) & ~(size_t)63;
        float* p = WSB + cur; cur += nf; return p;
    };
    auto allocB = [&](size_t ne) { return (__bf16*)alloc((ne + 1) / 2); };

    const size_t PADROWS = (size_t)(NTOT + 128) * 128;
    float* hA  = alloc(PADROWS);
    float* hB  = alloc(PADROWS);
    float* qb  = alloc(PADROWS);
    float* num = alloc((size_t)NTOT * 128);
    float* den = alloc((size_t)NTOT * 4);
    int*   mx  = (int*)alloc((size_t)NTOT * 4);
    __bf16* hAb = allocB(PADROWS);
    __bf16* hBb = allocB(PADROWS);
    __bf16* kbb = allocB(PADROWS);
    __bf16* vbb = allocB(PADROWS);
    __bf16* gbb = allocB(PADROWS);
    int Mpad[3]; __bf16* xb[3];
    for (int t = 0; t < 3; ++t) {
        Mpad[t] = (NN[t] + 127) & ~127;
        xb[t] = allocB((size_t)Mpad[t] * 32);
    }
    float* keb[4]; float* veb[4]; float* ae[4]; float* wex[4];
    for (int t = 0; t < 4; ++t) keb[t] = alloc((size_t)(NN[ET_SRC[t]] + 128) * 128);
    for (int t = 0; t < 4; ++t) veb[t] = alloc((size_t)(NN[ET_SRC[t]] + 128) * 128);
    for (int t = 0; t < 4; ++t) ae[t]  = alloc((size_t)nE[t] * 4);
    for (int t = 0; t < 4; ++t) wex[t] = alloc((size_t)nE[t] * 4);

    const int TB = 256;
    auto blks = [&](long n) { return (int)((n + TB - 1) / TB); };

    // ---- x -> bf16 (padded) ----
    for (int t = 0; t < 3; ++t)
        cvt_pad<<<blks((long)Mpad[t] * 32), TB, 0, stream>>>(
            X[t], xb[t], NN[t], INCH[t], Mpad[t]);

    // ---- embedding: h = x @ emb_w + emb_b (fp32 + bf16 mirror) ----
    for (int t = 0; t < 3; ++t) {
        gemm_n128<0, 1><<<(NN[t] + 127) / 128, 256, 0, stream>>>(
            xb[t], 32, F(EMB + 1 + 2 * t), F(EMB + 2 * t),
            hA + (size_t)GOFF[t] * 128, hAb + (size_t)GOFF[t] * 128,
            nullptr, nullptr, NN[t], INCH[t]);
    }

    // ---- two HGT conv layers ----
    float*  hcur  = hA;  float*  hnext  = hB;
    __bf16* hcurb = hAb; __bf16* hnextb = hBb;
    for (int layer = 0; layer < 2; ++layer) {
        const int CB = (layer == 0) ? CONV1 : CONV2;
        // K/Q/V projections (K,V -> bf16 only; Q -> fp32 only)
        for (int t = 0; t < 3; ++t) {
            const __bf16* src = hcurb + (size_t)GOFF[t] * 128;
            int g = (NN[t] + 127) / 128;
            gemm_n128<0, 4><<<g, 256, 0, stream>>>(src, 128, F(CB + 11 + 2 * t),
                F(CB + 10 + 2 * t), nullptr, kbb + (size_t)GOFF[t] * 128,
                nullptr, nullptr, NN[t], 128);
            gemm_n128<0, 4><<<g, 256, 0, stream>>>(src, 128, F(CB + 25 + 2 * t),
                F(CB + 24 + 2 * t), qb + (size_t)GOFF[t] * 128, nullptr,
                nullptr, nullptr, NN[t], 128);
            gemm_n128<0, 4><<<g, 256, 0, stream>>>(src, 128, F(CB + 34 + 2 * t),
                F(CB + 33 + 2 * t), nullptr, vbb + (size_t)GOFF[t] * 128,
                nullptr, nullptr, NN[t], 128);
        }
        // relation transforms: ke = k[src] x a_rel, ve = v[src] x m_rel
        for (int t = 0; t < 4; ++t) {
            int Ms = NN[ET_SRC[t]];
            dim3 g((Ms + 127) / 128, 4);
            relgemm<<<g, 256, 0, stream>>>(kbb + (size_t)GOFF[ET_SRC[t]] * 128,
                                           F(CB + 6 + t), keb[t], Ms);
            relgemm<<<g, 256, 0, stream>>>(vbb + (size_t)GOFF[ET_SRC[t]] * 128,
                                           F(CB + 16 + t), veb[t], Ms);
        }
        // segment softmax state init
        fill_f<<<blks((long)NTOT * 128), TB, 0, stream>>>(num, 0.0f, (long)NTOT * 128);
        fill_f<<<blks((long)NTOT * 4),   TB, 0, stream>>>(den, 0.0f, (long)NTOT * 4);
        fill_i<<<blks((long)NTOT * 4),   TB, 0, stream>>>(mx, (int)0x80000000, (long)NTOT * 4);
        // attention
        for (int t = 0; t < 4; ++t)
            attn_score<<<blks((long)nE[t] * 4), TB, 0, stream>>>(
                ei[t], nE[t], GOFF[ET_DST[t]], qb, keb[t], F(CB + 20 + t), ae[t], mx);
        for (int t = 0; t < 4; ++t)
            attn_ex<<<blks((long)nE[t] * 4), TB, 0, stream>>>(
                ei[t], nE[t], GOFF[ET_DST[t]], ae[t], mx, wex[t], den);
        for (int t = 0; t < 4; ++t)
            attn_scatter<<<blks((long)nE[t] * 128), TB, 0, stream>>>(
                ei[t], nE[t], GOFF[ET_DST[t]], wex[t], veb[t], num);
        // g = gelu(num/den) -> bf16
        gelu_div<<<blks((long)NTOT * 128), TB, 0, stream>>>(num, den, gbb, (long)NTOT * 128);
        // out = leaky_relu(sigmoid(skip)*(g@a_w + a_b) + (1-sigmoid(skip))*h)
        for (int t = 0; t < 3; ++t) {
            gemm_n128<1, 4><<<(NN[t] + 127) / 128, 256, 0, stream>>>(
                gbb + (size_t)GOFF[t] * 128, 128, F(CB + 1 + 2 * t), F(CB + 2 * t),
                hnext + (size_t)GOFF[t] * 128, hnextb + (size_t)GOFF[t] * 128,
                F(CB + 30 + t), hcur + (size_t)GOFF[t] * 128, NN[t], 128);
        }
        float* tf = hcur; hcur = hnext; hnext = tf;
        __bf16* tb = hcurb; hcurb = hnextb; hnextb = tb;
    }

    // ---- final head: out = h['order'] @ out_w + out_b ----
    out_head<<<blks(100000), TB, 0, stream>>>(
        hcur + (size_t)GOFF[1] * 128, F(OUTP + 1), F(OUTP), (float*)d_out, 100000);
}
